// LanguageMLP_42734924595240
// MI455X (gfx1250) — matrix operand order repaired and checked by reading, not compiled, not run on hardware
//
#include <hip/hip_runtime.h>
#include <hip/hip_bf16.h>
#include <math.h>

// ---------------------------------------------------------------------------
// Problem constants (from reference): B=8, T=128, E=64, P=128, F=8192, V=4096
// ---------------------------------------------------------------------------
#define BB 8
#define TT 128
#define EE 64
#define PP 128
#define FF 8192   // P*E
#define VV 4096
#define MM 1024   // B*T

#define BM 128
#define BN 128
#define BK 32

typedef __attribute__((ext_vector_type(16))) __bf16 v16bf;
typedef __attribute__((ext_vector_type(8)))  __bf16 v8bf;
typedef __attribute__((ext_vector_type(8)))  float  v8f;

// --------------------- CDNA5 data-mover feature detection -------------------
#if defined(__AMDGCN__) && \
    __has_builtin(__builtin_amdgcn_tensor_load_to_lds) && \
    __has_builtin(__builtin_amdgcn_s_wait_tensorcnt)
#define USE_TDM 1
#else
#define USE_TDM 0
#endif

#if defined(__AMDGCN__) && \
    __has_builtin(__builtin_amdgcn_global_load_async_to_lds_b128) && \
    __has_builtin(__builtin_amdgcn_s_wait_asynccnt)
#define USE_ASYNC_LDS 1
#else
#define USE_ASYNC_LDS 0
#endif

#if USE_ASYNC_LDS || USE_TDM
typedef int v4i_vec __attribute__((vector_size(16)));
typedef __attribute__((address_space(1))) v4i_vec* gptr_v4i;
typedef __attribute__((address_space(3))) v4i_vec* lptr_v4i;
typedef __attribute__((address_space(3))) void*    lds_vp;
#endif

#if USE_TDM
typedef unsigned int u32x4 __attribute__((vector_size(16)));
typedef int          i32x4 __attribute__((vector_size(16)));
typedef int          i32x8 __attribute__((vector_size(32)));

// Issue one TDM 2D tile load: tile_d1 rows x tile_d0 elements of bf16 from a
// row-major tensor (row pitch = tdim0 elements) into contiguous LDS.
// D# bit layout per CDNA5 ISA ch. 8 (groups 0/1; groups 2/3 zero for 2D).
__device__ __forceinline__ void tdm_load_tile_bf16(
    const __bf16* gsrc, __bf16* ldst,
    unsigned tdim0, unsigned tdim1, unsigned tile_d0, unsigned tile_d1)
{
    unsigned long g = (unsigned long)gsrc;                 // global byte address
    unsigned laddr  = (unsigned)(unsigned long)(lds_vp)(void*)ldst; // LDS byte addr

    u32x4 g0;
    g0[0] = 1u;                                            // count=1 (valid user D#)
    g0[1] = laddr;                                         // lds_addr
    g0[2] = (unsigned)(g & 0xffffffffu);                   // global_addr[31:0]
    g0[3] = (unsigned)((g >> 32) & 0x01ffffffu)            // global_addr[56:32]
          | (2u << 30);                                    // type=2 ("image")

    i32x8 g1;
    g1[0] = (int)(1u << 16);                               // data_size=1 -> 2 bytes
    g1[1] = (int)((tdim0 & 0xffffu) << 16);                // tensor_dim0[15:0]
    g1[2] = (int)(((tdim0 >> 16) & 0xffffu)                // tensor_dim0[31:16]
          |       ((tdim1 & 0xffffu) << 16));              // tensor_dim1[15:0]
    g1[3] = (int)(((tdim1 >> 16) & 0xffffu)                // tensor_dim1[31:16]
          |       ((tile_d0 & 0xffffu) << 16));            // tile_dim0
    g1[4] = (int)(tile_d1 & 0xffffu);                      // tile_dim1 (tile_dim2=0)
    g1[5] = (int)tdim0;                                    // tensor_dim0_stride[31:0]
    g1[6] = 0;                                             // stride hi / dim1_stride
    g1[7] = 0;

    i32x4 gz = {0, 0, 0, 0};                               // groups 2/3 unused (2D)

#if __clang_major__ >= 23
    i32x8 gz8 = {0, 0, 0, 0, 0, 0, 0, 0};
    __builtin_amdgcn_tensor_load_to_lds(g0, g1, gz, gz, gz8, 0);
#else
    __builtin_amdgcn_tensor_load_to_lds(g0, g1, gz, gz, 0);
#endif
}
#endif // USE_TDM

__device__ __forceinline__ float gelu_tanh_f(float x) {
    float x3 = x * x * x;
    return 0.5f * x * (1.0f + tanhf(0.7978845608028654f * (x + 0.044715f * x3)));
}

// A-fragment (ISA 16-bit A 16x32 layout): elems 0..7 = K kbase..kbase+7,
// elems 8..15 = K kbase+16..kbase+23  -> two 16B LDS chunks at p and p+16.
__device__ __forceinline__ v16bf ld_frag_a(const __bf16* p) {
    v8bf lo = *(const v8bf*)p;
    v8bf hi = *(const v8bf*)(p + 16);
    return __builtin_shufflevector(lo, hi, 0,1,2,3,4,5,6,7,8,9,10,11,12,13,14,15);
}

// B-fragment: elems 0..7 = K kbase..kbase+7, elems 8..15 = K kbase+16..kbase+23
__device__ __forceinline__ v16bf ld_frag_b(const __bf16* p) {
    v8bf lo = *(const v8bf*)p;
    v8bf hi = *(const v8bf*)(p + 16);
    return __builtin_shufflevector(lo, hi, 0,1,2,3,4,5,6,7,8,9,10,11,12,13,14,15);
}

// ---------------------------------------------------------------------------
// 1) Embedding + causal sliding-window gather -> X[MM][FF] in bf16
// ---------------------------------------------------------------------------
__global__ __launch_bounds__(256) void embed_gather_kernel(
    const int* __restrict__ ids, const float* __restrict__ wte,
    __bf16* __restrict__ X)
{
    long idx = (long)blockIdx.x * 256 + threadIdx.x;   // over MM*FF
    int e =  idx & (EE - 1);
    int j = (idx >> 6) & (PP - 1);
    int t = (idx >> 13) & (TT - 1);
    int b =  idx >> 20;
    int src = j - (PP - 1) + t;
    float v = 0.0f;
    if (src >= 0) {
        int tok = ids[b * TT + src];
        v = wte[(long)tok * EE + e];
    }
    X[idx] = (__bf16)v;
}

// ---------------------------------------------------------------------------
// 2) f32 -> bf16 weight conversion
// ---------------------------------------------------------------------------
__global__ __launch_bounds__(256) void cvt_bf16_kernel(
    const float* __restrict__ in, __bf16* __restrict__ out, long n)
{
    long i = ((long)blockIdx.x * 256 + threadIdx.x) * 4;
    if (i + 3 < n) {
        float4 f = *(const float4*)(in + i);
        out[i + 0] = (__bf16)f.x;
        out[i + 1] = (__bf16)f.y;
        out[i + 2] = (__bf16)f.z;
        out[i + 3] = (__bf16)f.w;
    }
}

// ---------------------------------------------------------------------------
// Per-lane tile staging fallback (async global->LDS if available, else sync)
// ---------------------------------------------------------------------------
__device__ __forceinline__ void stage_tiles_lanes(
    const __bf16* __restrict__ A, const __bf16* __restrict__ W,
    __bf16* As, __bf16* Bs,
    int tid, int blockM, int blockN, int K, int k0)
{
    #pragma unroll
    for (int it = 0; it < 2; ++it) {
        int s   = tid + it * 256;
        int row = s >> 2;
        int q   = (s & 3) * 8;
#if USE_ASYNC_LDS
        __builtin_amdgcn_global_load_async_to_lds_b128(
            (gptr_v4i)&A[(long)(blockM + row) * K + k0 + q],
            (lptr_v4i)&As[row * BK + q], 0, 0);
        __builtin_amdgcn_global_load_async_to_lds_b128(
            (gptr_v4i)&W[(long)(blockN + row) * K + k0 + q],
            (lptr_v4i)&Bs[row * BK + q], 0, 0);
#else
        *(uint4*)&As[row * BK + q] = *(const uint4*)&A[(long)(blockM + row) * K + k0 + q];
        *(uint4*)&Bs[row * BK + q] = *(const uint4*)&W[(long)(blockN + row) * K + k0 + q];
#endif
    }
}

// ---------------------------------------------------------------------------
// 3) NT GEMM:  C[M][N] = act(A[M][K] * W[N][K]^T + bias[N])
//    bf16 in, f32 WMMA accumulate. 256 threads = 8 waves (wave32), waves 4x2,
//    each wave owns a 32x64 tile = 2x4 v_wmma_f32_16x16x32_bf16 accumulators.
//    Double-buffered LDS; tiles staged by the Tensor Data Mover (one D# per
//    tile, issued by wave 0, pipelined on TENSORcnt), falling back to
//    per-lane ASYNCcnt copies.
// ---------------------------------------------------------------------------
template <bool GELU, typename OUT_T>
__global__ __launch_bounds__(256) void gemm_bf16_kernel(
    const __bf16* __restrict__ A,   // [M][K] row-major
    const __bf16* __restrict__ W,   // [N][K] row-major (i.e. B^T)
    const float*  __restrict__ bias,// [N]
    OUT_T* __restrict__ C,          // [M][N]
    int M, int N, int K)
{
    __shared__ __align__(16) __bf16 As[2][BM * BK];
    __shared__ __align__(16) __bf16 Bs[2][BN * BK];

    const int tid  = threadIdx.x;
    const int lane = tid & 31;
    const int wave = tid >> 5;        // 0..7
    const int wm   = wave & 3;        // 0..3 : 32-row band
    const int wn   = wave >> 2;       // 0..1 : 64-col band
    const int blockM = blockIdx.y * BM;
    const int blockN = blockIdx.x * BN;

    const int lrow = lane & 15;       // row/col within 16x16 subtile
    const int kba  = (lane >> 4) * 8; // A fragment K base per ISA layout
    const int kbb  = (lane >> 4) * 8; // B fragment K base per ISA layout

    v8f acc[2][4];
    v8f vzero = {0.f, 0.f, 0.f, 0.f, 0.f, 0.f, 0.f, 0.f};
    #pragma unroll
    for (int i = 0; i < 2; ++i)
        #pragma unroll
        for (int j = 0; j < 4; ++j)
            acc[i][j] = vzero;

    // ------------------ prologue: stage first K-tile into buffer 0
#if USE_TDM
    if (wave == 0) {
        tdm_load_tile_bf16(&A[(long)blockM * K], As[0], (unsigned)K, (unsigned)M, BK, BM);
        tdm_load_tile_bf16(&W[(long)blockN * K], Bs[0], (unsigned)K, (unsigned)N, BK, BN);
    }
#else
    stage_tiles_lanes(A, W, As[0], Bs[0], tid, blockM, blockN, K, 0);
#endif

    int buf = 0;
    for (int k0 = 0; k0 < K; k0 += BK) {
        const bool has_next = (k0 + BK < K);   // uniform

#if USE_TDM
        // wave 0 drives the Tensor Data Mover; TENSORcnt completes in order,
        // so outstanding <= 2 (the next tile's two D#s) => current tile landed.
        if (wave == 0) {
            if (has_next) {
                tdm_load_tile_bf16(&A[(long)blockM * K + (k0 + BK)], As[buf ^ 1],
                                   (unsigned)K, (unsigned)M, BK, BM);
                tdm_load_tile_bf16(&W[(long)blockN * K + (k0 + BK)], Bs[buf ^ 1],
                                   (unsigned)K, (unsigned)N, BK, BN);
                __builtin_amdgcn_s_wait_tensorcnt(2);
            } else {
                __builtin_amdgcn_s_wait_tensorcnt(0);
            }
        }
#else
        if (has_next)
            stage_tiles_lanes(A, W, As[buf ^ 1], Bs[buf ^ 1], tid, blockM, blockN, K, k0 + BK);
#if USE_ASYNC_LDS
        // async copies complete in order: outstanding <= 4 (this thread's 4
        // next-tile copies) implies the current tile's 4 copies have landed.
        if (has_next) __builtin_amdgcn_s_wait_asynccnt(4);
        else          __builtin_amdgcn_s_wait_asynccnt(0);
#endif
#endif
        __syncthreads();   // current tile visible to all 8 waves

        const __bf16* as = As[buf];
        const __bf16* bs = Bs[buf];

        v16bf af[2];
        #pragma unroll
        for (int i = 0; i < 2; ++i)
            af[i] = ld_frag_a(&as[(wm * 32 + i * 16 + lrow) * BK + kba]);

        v16bf bfr[4];
        #pragma unroll
        for (int j = 0; j < 4; ++j)
            bfr[j] = ld_frag_b(&bs[(wn * 64 + j * 16 + lrow) * BK + kbb]);

        #pragma unroll
        for (int i = 0; i < 2; ++i)
            #pragma unroll
            for (int j = 0; j < 4; ++j)
                acc[i][j] = __builtin_amdgcn_wmma_f32_16x16x32_bf16(
                    false, af[i], false, bfr[j], (short)0, acc[i][j], false, false);

        __syncthreads();   // done reading buf before it is re-staged
        buf ^= 1;
    }

    // epilogue: bias (+gelu). C/D layout: elem r -> row r + (lane>=16 ? 8 : 0)
    const int rsel = (lane >> 4) * 8;
    #pragma unroll
    for (int i = 0; i < 2; ++i) {
        int rowb = blockM + wm * 32 + i * 16 + rsel;
        #pragma unroll
        for (int j = 0; j < 4; ++j) {
            int col  = blockN + wn * 64 + j * 16 + lrow;
            float bv = bias[col];
            #pragma unroll
            for (int r = 0; r < 8; ++r) {
                float v = acc[i][j][r] + bv;
                if (GELU) v = gelu_tanh_f(v);
                C[(long)(rowb + r) * N + col] = (OUT_T)v;
            }
        }
    }
}

// ---------------------------------------------------------------------------
// Launcher
// ---------------------------------------------------------------------------
extern "C" void kernel_launch(void* const* d_in, const int* in_sizes, int n_in,
                              void* d_out, int out_size, void* d_ws, size_t ws_size,
                              hipStream_t stream) {
    (void)in_sizes; (void)n_in; (void)out_size; (void)ws_size;

    const int*   ids   = (const int*)  d_in[0];
    const float* wte   = (const float*)d_in[1];
    const float* fc1_w = (const float*)d_in[2];
    const float* fc1_b = (const float*)d_in[3];
    const float* fc2_w = (const float*)d_in[4];
    const float* fc2_b = (const float*)d_in[5];
    float* y = (float*)d_out;

    // workspace layout (224 MB total)
    char* ws = (char*)d_ws;
    __bf16* X  = (__bf16*)(ws);                              // 1024*8192*2  = 16 MB
    __bf16* H  = (__bf16*)(ws + (size_t)16  * 1024 * 1024);  // 16 MB
    __bf16* W1 = (__bf16*)(ws + (size_t)32  * 1024 * 1024);  // 8192*8192*2  = 128 MB
    __bf16* W2 = (__bf16*)(ws + (size_t)160 * 1024 * 1024);  // 4096*8192*2  = 64 MB

    // 1) embedding gather -> X (bf16)
    {
        long n = (long)MM * FF;                  // 8,388,608
        embed_gather_kernel<<<(int)(n / 256), 256, 0, stream>>>(ids, wte, X);
    }
    // 2) weight conversions f32 -> bf16 (both then live in the 192MB L2)
    {
        long n1 = (long)FF * FF;                 // 67,108,864
        cvt_bf16_kernel<<<(int)(n1 / (256 * 4)), 256, 0, stream>>>(fc1_w, W1, n1);
        long n2 = (long)VV * FF;                 // 33,554,432
        cvt_bf16_kernel<<<(int)(n2 / (256 * 4)), 256, 0, stream>>>(fc2_w, W2, n2);
    }
    // 3) H = gelu(X @ W1^T + b1)   [1024 x 8192], K = 8192
    {
        dim3 grid(FF / BN, MM / BM);             // (64, 8)
        gemm_bf16_kernel<true, __bf16><<<grid, 256, 0, stream>>>(
            X, W1, fc1_b, H, MM, FF, FF);
    }
    // 4) Y = H @ W2^T + b2         [1024 x 4096], K = 8192
    {
        dim3 grid(VV / BN, MM / BM);             // (32, 8)
        gemm_bf16_kernel<false, float><<<grid, 256, 0, stream>>>(
            H, W2, fc2_b, y, MM, VV, FF);
    }
}
